// _CrossNeuronBlock_20083267076587
// MI455X (gfx1250) — matrix operand order repaired and checked by reading, not compiled, not run on hardware
//
#include <hip/hip_runtime.h>

// CDNA5 / gfx1250: wave32, V_WMMA_F32_16X16X32_F16 + GLOBAL_LOAD_ASYNC_TO_LDS_B128.
typedef __attribute__((ext_vector_type(16))) _Float16 v16h;
typedef __attribute__((ext_vector_type(8)))  _Float16 v8h;
typedef __attribute__((ext_vector_type(8)))  float    v8f;

static constexpr int Bn = 32;     // batch
static constexpr int Cn = 512;    // channels
static constexpr int HWn = 1024;  // spatial

// ---------------------------------------------------------------------------
// Kernel 1: per-(b,c) channel mean over 1024 spatial elements.
__global__ void k_means(const float* __restrict__ x, float* __restrict__ xm) {
  const int wave = threadIdx.x >> 5;
  const int lane = threadIdx.x & 31;
  const int row  = blockIdx.x * 8 + wave;            // [0, Bn*Cn)
  const float4* p = reinterpret_cast<const float4*>(x + (size_t)row * HWn);
  float s = 0.f;
#pragma unroll
  for (int i = 0; i < 8; ++i) {
    float4 v = p[lane + 32 * i];
    s += v.x + v.y + v.z + v.w;
  }
#pragma unroll
  for (int off = 16; off > 0; off >>= 1) s += __shfl_xor(s, off, 32);
  if (lane == 0) xm[row] = s * (1.0f / (float)HWn);
}

// ---------------------------------------------------------------------------
// Kernel 2: transpose + convert x[b][c][s] (f32) -> xT[b][s][c] (f16).
__global__ void k_transpose(const float* __restrict__ x, _Float16* __restrict__ xT) {
  __shared__ _Float16 tile[32][33];
  const int b  = blockIdx.z;
  const int c0 = blockIdx.y * 32;
  const int s0 = blockIdx.x * 32;
  const int tx = threadIdx.x, ty = threadIdx.y;
#pragma unroll
  for (int r = 0; r < 4; ++r) {
    const int c = c0 + ty + r * 8;
    tile[ty + r * 8][tx] = (_Float16)x[((size_t)b * Cn + c) * HWn + (s0 + tx)];
  }
  __syncthreads();
#pragma unroll
  for (int r = 0; r < 4; ++r) {
    const int s = s0 + ty + r * 8;
    xT[((size_t)b * HWn + s) * Cn + (c0 + tx)] = tile[tx][ty + r * 8];
  }
}

// ---------------------------------------------------------------------------
// Kernel 3: attnT[b][j][k] = exp(-(m_j-m_k)^2) / sum_k exp(-(m_j-m_k)^2)  (f16)
// Softmax max is 0 (k=j term), so plain exp is stable.
__global__ void k_attn(const float* __restrict__ xm, _Float16* __restrict__ attnT) {
  __shared__ float m[Cn];
  const int b = blockIdx.x;
  const int quarter = blockIdx.y;
  for (int i = threadIdx.x; i < Cn; i += blockDim.x) m[i] = xm[b * Cn + i];
  __syncthreads();
  const int wave = threadIdx.x >> 5;
  const int lane = threadIdx.x & 31;
  for (int jj = 0; jj < 16; ++jj) {
    const int j = quarter * 128 + jj * 8 + wave;
    const float mj = m[j];
    float s = 0.f;
    for (int k = lane; k < Cn; k += 32) {
      const float d = mj - m[k];
      s += __expf(-d * d);
    }
#pragma unroll
    for (int off = 16; off > 0; off >>= 1) s += __shfl_xor(s, off, 32);
    const float inv = 1.0f / s;
    _Float16* row = attnT + ((size_t)b * Cn + j) * Cn;
    for (int k = lane; k < Cn; k += 32) {
      const float d = mj - m[k];
      row[k] = (_Float16)(__expf(-d * d) * inv);
    }
  }
}

// ---------------------------------------------------------------------------
// Kernel 4: batched GEMM out_t[b] = attnT[b] (512x512) @ X[b] (512x1024),
// fused relu(x + 0.1*out_t).
// Block = 8 waves covering 8 j-tiles that SHARE one 64-column B tile:
//   - B tile (64 s-rows x 512 k, f16, +8-half row pad) staged into LDS with
//     GLOBAL_LOAD_ASYNC_TO_LDS_B128 (ASYNCcnt), then read as ds_load_b128.
//     Row stride 1040B -> lane n hits banks 4n..4n+3: conflict-free.
//   - A fragments streamed from global (16 contiguous rows of attnT per wave).
// A/B fragments use contiguous k order (element s of lane half hi = k 16*hi + s) in both, not the ISA table order.
__global__ void __launch_bounds__(256) k_gemm(const float* __restrict__ x,
                                              const _Float16* __restrict__ attnT,
                                              const _Float16* __restrict__ xT,
                                              float* __restrict__ out) {
  constexpr int PAD = 8;                // halves (16 B) of row padding
  constexpr int LDB = Cn + PAD;         // 520 halves per staged s-row
  __shared__ _Float16 tileB[64 * LDB];

  const int tid  = threadIdx.x;
  const int wave = tid >> 5;
  const int lane = tid & 31;

  // block -> (b, jg, sg): 32 * 4 * 16 = 2048 blocks
  const int blk = blockIdx.x;
  const int b   = blk >> 6;
  const int jg  = (blk >> 4) & 3;
  const int sg  = blk & 15;
  const int s0  = sg * 64;

  // ---- async-stage B tile: xT[b][s0..s0+63][0..511] -> LDS (16 chunks/thread)
  {
    const _Float16* src = xT + ((size_t)b * HWn + s0) * Cn;
#pragma unroll
    for (int q = tid; q < 4096; q += 256) {
      const int row = q >> 6;           // 0..63 (s-row)
      const int col = (q & 63) * 8;     // halves within row
      const uint64_t g = (uint64_t)(uintptr_t)(src + (size_t)row * Cn + col);
      const uint32_t l = (uint32_t)(uintptr_t)(&tileB[row * LDB + col]);
      asm volatile("global_load_async_to_lds_b128 %0, %1, off"
                   :: "v"(l), "v"(g) : "memory");
    }
    asm volatile("s_wait_asynccnt 0" ::: "memory");
    __syncthreads();
  }

  const int jt = jg * 8 + wave;         // this wave's j tile (0..31)
  const int j0 = jt * 16;
  const int n  = lane & 15;
  const int hi = lane >> 4;

  // A: row j0+n of attnT; per K-step lane holds halves K[hi*16..+15]
  const _Float16* Arow = attnT + ((size_t)b * Cn + (j0 + n)) * Cn;

  union V16 { v16h v; struct { v8h lo, hi; } p; };
  v8f acc[4] = {};

  for (int k0 = 0; k0 < Cn; k0 += 32) {
    __builtin_prefetch(Arow + k0 + 64, 0, 0);        // global_prefetch_b8
    V16 a;
    a.p.lo = *reinterpret_cast<const v8h*>(Arow + k0 + hi * 16);
    a.p.hi = *reinterpret_cast<const v8h*>(Arow + k0 + hi * 16 + 8);
#pragma unroll
    for (int sub = 0; sub < 4; ++sub) {
      // B: column n of sub-tile = staged s-row; lane holds halves K[k0+hi*16..+15]
      const _Float16* bp = &tileB[(sub * 16 + n) * LDB + k0 + hi * 16];
      V16 bf;
      bf.p.lo = *reinterpret_cast<const v8h*>(bp);
      bf.p.hi = *reinterpret_cast<const v8h*>(bp + 8);
      acc[sub] = __builtin_amdgcn_wmma_f32_16x16x32_f16(
          /*neg_a=*/false, a.v, /*neg_b=*/false, bf.v,
          /*c_mod=*/(short)0, acc[sub], /*reuse_a=*/false, /*reuse_b=*/false);
    }
  }

  // Epilogue: D VGPR r -> (j = j0 + r + 8*hi, s = s0 + sub*16 + n)
#pragma unroll
  for (int sub = 0; sub < 4; ++sub) {
#pragma unroll
    for (int r = 0; r < 8; ++r) {
      const int j = j0 + r + hi * 8;
      const int s = s0 + sub * 16 + n;
      const size_t idx = ((size_t)b * Cn + j) * HWn + s;
      const float v = x[idx] + 0.1f * acc[sub][r];
      out[idx] = v > 0.f ? v : 0.f;
    }
  }
}

// ---------------------------------------------------------------------------
extern "C" void kernel_launch(void* const* d_in, const int* in_sizes, int n_in,
                              void* d_out, int out_size, void* d_ws, size_t ws_size,
                              hipStream_t stream) {
  const float* x = (const float*)d_in[0];
  float* out = (float*)d_out;

  // workspace: [xm: 64KB][attnT: 16MB][xT: 32MB]  (~48MB total)
  char* ws = (char*)d_ws;
  float*    xm    = (float*)ws;
  _Float16* attnT = (_Float16*)(ws + 64 * 1024);
  _Float16* xT    = (_Float16*)(ws + 64 * 1024 + (size_t)Bn * Cn * Cn * sizeof(_Float16));

  k_means    <<<dim3((Bn * Cn) / 8), dim3(256), 0, stream>>>(x, xm);
  k_transpose<<<dim3(HWn / 32, Cn / 32, Bn), dim3(32, 8), 0, stream>>>(x, xT);
  k_attn     <<<dim3(Bn, 4), dim3(256), 0, stream>>>(xm, attnT);
  k_gemm     <<<dim3(Bn * 4 * 16), dim3(256), 0, stream>>>(x, attnT, xT, out);
}
